// ProcessFeatures_2362232012968
// MI455X (gfx1250) — matrix operand order repaired and checked by reading, not compiled, not run on hardware
//
#include <hip/hip_runtime.h>
#include <hip/hip_bf16.h>

// ---------- types ----------
typedef __attribute__((ext_vector_type(16))) __bf16        v16bf;
typedef __attribute__((ext_vector_type(8)))  float         v8f;
typedef __attribute__((ext_vector_type(4)))  unsigned int  u32x4;
typedef __attribute__((ext_vector_type(8)))  int           i32x8;
typedef __attribute__((ext_vector_type(4)))  int           i32x4;

// Problem constants (from reference): sat/grd (128, 4, 64, 16) fp32
#define NB   128            // batch (both sat and grd)
#define NH   4
#define NW   64
#define NC   16
#define NK   4096           // h*w*c flattened K
#define NELEM (NB*NK)       // 524288 per tensor
// LDS row stride: 4096 bf16 + 16B pad = 8208 B = 2052 dwords == 4 (mod 64 banks)
// -> each of the 16 rows maps to a disjoint 4-dword bank group per half-wave phase.
#define ROWB 8208
#define ROWE 4104           // row stride in bf16 elements
#define GRD_LDS_OFF (16u * ROWB)       // 131328
#define LDS_BYTES   (2u * GRD_LDS_OFF) // 262656 <= 320KB

// float -> bf16, round to nearest even
static __device__ __forceinline__ unsigned short f2bf(float f) {
  unsigned int b = __float_as_uint(f);
  unsigned int r = b + 0x7FFFu + ((b >> 16) & 1u);
  return (unsigned short)(r >> 16);
}

// ---------- kernel 1: copy inputs to output + bf16 conversion ----------
__global__ void cv_prep_kernel(const float* __restrict__ sat,
                               const float* __restrict__ grd,
                               float* __restrict__ out_sat,
                               float* __restrict__ out_grd,
                               unsigned short* __restrict__ satb,
                               unsigned short* __restrict__ grdb) {
  int i = blockIdx.x * blockDim.x + threadIdx.x;
  if (i < NELEM) {
    float s = sat[i], g = grd[i];
    out_sat[i] = s;
    out_grd[i] = g;
    satb[i] = f2bf(s);
    grdb[i] = f2bf(g);
  }
}

// ---------- kernel 2: per-batch Frobenius norm of sat ----------
__global__ void cv_norm_kernel(const float* __restrict__ sat,
                               float* __restrict__ nrm) {
  __shared__ float red[256];
  const float* row = sat + (size_t)blockIdx.x * NK;
  float s = 0.f;
  for (int i = threadIdx.x; i < NK; i += 256) {
    float v = row[i];
    s += v * v;
  }
  red[threadIdx.x] = s;
  __syncthreads();
  for (int off = 128; off > 0; off >>= 1) {
    if ((int)threadIdx.x < off) red[threadIdx.x] += red[threadIdx.x + off];
    __syncthreads();
  }
  if (threadIdx.x == 0) nrm[blockIdx.x] = sqrtf(red[0]);
}

// ---------- TDM: load 16 rows x 4096 bf16 into LDS with 16B row padding ----------
// Iterate mode (CDNA5 ISA): tile = 1 row of 4096 elems,
// 16 iterations; global_addr += 4096 elems/iter, lds_addr += 4104 elems (8208 B)
// per iter -> bank-conflict-free padded rows. One descriptor per tile.
static __device__ __forceinline__ void tdm_load_tile(unsigned long long gaddr,
                                                     unsigned int lds_off) {
  u32x4 g0;
  g0[0] = 1u;                                             // count=1, user desc
  g0[1] = lds_off;                                        // lds_addr (bytes)
  g0[2] = (unsigned int)(gaddr & 0xffffffffu);            // global_addr[31:0]
  g0[3] = (unsigned int)((gaddr >> 32) & 0x1ffffffu)      // global_addr[56:32]
        | (2u << 30);                                     // type = 2 ("image")
  i32x8 g1;
  g1[0] = (int)((1u << 16) | (1u << 19)); // mask=0, data_size=2B, iterate_enable=1
  g1[1] = (int)(4096u << 16);             // tensor_dim0[15:0] in [31:16]
  g1[2] = (int)(128u  << 16);             // tensor_dim0 hi=0 | tensor_dim1 lo=128
  g1[3] = (int)(4096u << 16);             // tensor_dim1 hi=0 | tile_dim0=4096
  g1[4] = 1;                              // tile_dim1=1 (one row/iter), tile_dim2=0
  g1[5] = 4096;                           // tensor_dim0_stride low
  g1[6] = 0;
  g1[7] = 0;
  i32x4 g2;
  g2[0] = 0;                              // tensor_dim2
  g2[1] = (int)ROWE;                      // lds_addr_increment (data_size units)
  g2[2] = 4096;                           // global_addr_increment (data_size units)
  g2[3] = (int)(15u << 16);               // iterate_count=15 -> 16 iterations
  i32x4 z4 = {0, 0, 0, 0};
  i32x8 z8 = {0, 0, 0, 0, 0, 0, 0, 0};
  __builtin_amdgcn_tensor_load_to_lds(g0, g1, g2, z4, z8, 0);
}

// ---------- kernel 3: all-shift correlation via bf16 WMMA ----------
// grid = (8 bs-tiles, 8 bg-tiles, 8 shift-groups); 128 threads = 4 waves;
// each wave owns 2 shifts. corr layout: [bs][bg][s] (s contiguous).
__global__ __launch_bounds__(128) void cv_corr_wmma_kernel(
    const unsigned short* __restrict__ satb,
    const unsigned short* __restrict__ grdb,
    float* __restrict__ corr) {
  extern __shared__ unsigned char smem[];
  const int bs0 = blockIdx.x * 16;
  const int bg0 = blockIdx.y * 16;
  const int tid  = threadIdx.x;
  const int wave = tid >> 5;
  const int lane = tid & 31;

  // --- fill LDS tiles with the Tensor Data Mover (wave 0 issues, WG barrier publishes)
  if (wave == 0) {
    tdm_load_tile((unsigned long long)(uintptr_t)(satb + (size_t)bs0 * NK), 0u);
    tdm_load_tile((unsigned long long)(uintptr_t)(grdb + (size_t)bg0 * NK), GRD_LDS_OFF);
    __builtin_amdgcn_s_wait_tensorcnt(0);
  }
  __syncthreads();

  const int hi = lane >> 4;          // 0/1 half-wave
  const int r  = lane & 15;          // row within tile (bs for A, bg for B)
  const int kbA = hi * 8;            // A lane K offset per ISA layout
  const int kbB = hi * 8;            // B lane K offset per ISA layout
  const int s0 = blockIdx.z * 8 + wave * 2;
  const int s1 = s0 + 1;

  const uint4* sa = (const uint4*)(smem + (size_t)r * ROWB);               // sat row r
  const uint4* gb = (const uint4*)(smem + GRD_LDS_OFF + (size_t)r * ROWB); // grd row r

  v8f acc0 = {};
  v8f acc1 = {};

  for (int k = 0; k < NK; k += 32) {
    union { v16bf v; uint4 u[2]; } bf, a0, a1;
    // B: K = k+kbB .. k+kbB+7, then k+kbB+16 .. k+kbB+23
    int bidx = (k + kbB) >> 3;
    bf.u[0] = gb[bidx];
    bf.u[1] = gb[bidx + 2];
    // A: circular shift applied in LDS index. Element idx = (h*64+w)*16 + c.
    int kk   = k + kbA;              // chunk0 start; chunk1 = kk+16 (same h)
    int h    = kk >> 10;
    int dw   = (kk >> 4) & 63;
    int base = (h << 7) + hi;        // (h*64*16 + c)/8 in uint4 units, c = 8*hi
    int d00 = (dw + s0) & 63, d10 = (dw + 1 + s0) & 63;
    int d01 = (dw + s1) & 63, d11 = (dw + 1 + s1) & 63;
    a0.u[0] = sa[base + d00 * 2];
    a0.u[1] = sa[base + d10 * 2];
    a1.u[0] = sa[base + d01 * 2];
    a1.u[1] = sa[base + d11 * 2];

    acc0 = __builtin_amdgcn_wmma_f32_16x16x32_bf16(false, a0.v, false, bf.v,
                                                   (short)0, acc0, false, false);
    acc1 = __builtin_amdgcn_wmma_f32_16x16x32_bf16(false, a1.v, false, bf.v,
                                                   (short)0, acc1, false, false);
  }

  // C/D layout: lanes 0-15 -> N=lane, M=v; lanes 16-31 -> N=lane-16, M=v+8
  const int bg = bg0 + r;
  for (int v = 0; v < 8; ++v) {
    int bs = bs0 + v + hi * 8;
    size_t o = ((size_t)(bs * NB + bg)) * NW;
    corr[o + s0] = acc0[v];
    corr[o + s1] = acc1[v];
  }
}

// ---------- kernel 4: argmax over shifts -> orien, distance ----------
__global__ void cv_finalize_kernel(const float* __restrict__ corr,
                                   const float* __restrict__ nrm,
                                   float* __restrict__ dist,
                                   float* __restrict__ orien) {
  int t = blockIdx.x * blockDim.x + threadIdx.x;   // 16384 (bs, bg) pairs
  if (t >= NB * NB) return;
  int bs = t >> 7;
  int bg = t & 127;
  const float* c = corr + (size_t)t * NW;          // contiguous 64 shifts
  float vmax = c[0];
  int smax = 0;
  for (int s = 1; s < NW; ++s) {
    float v = c[s];
    if (v > vmax) { vmax = v; smax = s; }          // first-max, matches argmax
  }
  float n = fmaxf(nrm[bs], 1e-12f);
  dist[bg * NB + bs]  = 2.0f - 2.0f * (vmax / n);  // distance = (2 - 2*dot).T
  orien[bs * NB + bg] = (float)smax;
}

// ---------- launch ----------
extern "C" void kernel_launch(void* const* d_in, const int* in_sizes, int n_in,
                              void* d_out, int out_size, void* d_ws, size_t ws_size,
                              hipStream_t stream) {
  const float* sat = (const float*)d_in[0];
  const float* grd = (const float*)d_in[1];
  float* out = (float*)d_out;

  unsigned char* ws = (unsigned char*)d_ws;
  unsigned short* satb = (unsigned short*)(ws);                 // 1 MB
  unsigned short* grdb = (unsigned short*)(ws + (1u << 20));    // 1 MB
  float* corr = (float*)(ws + (2u << 20));                      // 4 MB
  float* nrm  = (float*)(ws + (6u << 20));                      // 512 B

  float* out_sat  = out;                    // 524288
  float* out_grd  = out + NELEM;            // 524288
  float* out_dist = out + 2 * NELEM;        // 16384
  float* out_ori  = out + 2 * NELEM + NB * NB;

  cv_prep_kernel<<<(NELEM + 255) / 256, 256, 0, stream>>>(sat, grd, out_sat, out_grd,
                                                          satb, grdb);
  cv_norm_kernel<<<NB, 256, 0, stream>>>(sat, nrm);
  cv_corr_wmma_kernel<<<dim3(8, 8, 8), 128, LDS_BYTES, stream>>>(satb, grdb, corr);
  cv_finalize_kernel<<<(NB * NB + 255) / 256, 256, 0, stream>>>(corr, nrm, out_dist,
                                                                out_ori);
}
